// SimpleGatedGCNNet_65189013618922
// MI455X (gfx1250) — matrix operand order repaired and checked by reading, not compiled, not run on hardware
//
#include <hip/hip_runtime.h>
#include <stdint.h>

// ---------------------------------------------------------------------------
// Gated-GCN forward for MI455X (gfx1250).
//  - all dense GEMMs on v_wmma_f32_16x16x32_f16 (f32 accumulate)
//  - f16 shadow copies of activations halve the dominant HBM read streams and
//    let A tiles be staged with global_load_async_to_lds_b128 (per-lane
//    addresses -> graph gathers ride the async path for free)
//  - BN column stats fused into producing GEMM epilogues (no extra passes)
//  - segment-sum via f32 atomics (num/den are L2-resident: 12.8 MB << 192 MB)
// ---------------------------------------------------------------------------

static constexpr int NN = 25000;    // nodes
static constexpr int EE = 400000;   // edges
static constexpr int DD = 64;       // hidden dim
static constexpr int NLAYERS = 4;

typedef __attribute__((ext_vector_type(16))) _Float16 v16h;
typedef __attribute__((ext_vector_type(8)))  _Float16 v8h;
typedef __attribute__((ext_vector_type(4)))  _Float16 v4h;
typedef __attribute__((ext_vector_type(2)))  _Float16 v2h;
typedef __attribute__((ext_vector_type(8)))  float    v8f;

enum AMode { A_DIRECT = 0, A_GATHER = 1, A_FEAT = 2 };

// 16B global -> LDS async copy (CDNA5).  lds_off is the wave-relative LDS
// byte address (low 32 bits of a generic pointer to __shared__).
__device__ __forceinline__ void async_ld16(uint32_t lds_off, const void* g) {
    asm volatile("global_load_async_to_lds_b128 %0, %1, off"
                 :: "v"(lds_off), "v"((uint64_t)(uintptr_t)g)
                 : "memory");
}
__device__ __forceinline__ void async_wait0() {
    asm volatile("s_wait_asynccnt 0" ::: "memory");
}

// ---------------------------------------------------------------------------
// WMMA GEMM core: Y[M x 64] = act( A[M x K] @ W[K x 64] + bias (+ gathers) )
// 128 rows x 64 cols per block of 256 threads (8 wave32 waves); each wave
// owns a 16x64 strip = 4x v_wmma_f32_16x16x32_f16 per 32-wide K-step.
// Optional fused column sum/sumsq (STATS) for train-mode BN.
// ---------------------------------------------------------------------------
template<int AMODE, bool SRC16, bool RELU, bool ACCUM, bool GATHER2,
         bool WRITE16, bool STATS>
__global__ __launch_bounds__(256)
void gemm64_kernel(const float* __restrict__ X, int ldx,          // f32 source
                   const _Float16* __restrict__ Xh,               // f16 source (ldx in halfs)
                   const int* __restrict__ gidx,
                   const _Float16* __restrict__ fe16,             // A_FEAT: e (f16)
                   const _Float16* __restrict__ fh16,             // A_FEAT: h (f16)
                   const int* __restrict__ src, const int* __restrict__ dst,
                   const int* __restrict__ revmap,
                   const float* __restrict__ W, int ldw,
                   const float* __restrict__ bias,
                   const float* __restrict__ G0, const float* __restrict__ G1,
                   float* __restrict__ Y, int ldy,
                   _Float16* __restrict__ Y16,
                   float* __restrict__ statsOut,
                   int M, int K)
{
    __shared__ _Float16 As[128][72];   // A tile: 128 rows x 64 halfs (+pad)
    __shared__ _Float16 Wt[64][72];    // W tile transposed: Wt[n][k] (+pad)

    const int t        = threadIdx.x;
    const int blockRow = blockIdx.x * 128;
    const int lane     = t & 31;
    const int m0       = (t >> 5) * 16;   // wave's row strip within the tile

    const uint32_t as_base = (uint32_t)(uintptr_t)&As[0][0];

    // partial block (nodes only): pre-zero OOB rows once; async lanes for
    // those rows are exec-masked off and never touch them again.
    if (SRC16) {
        int zs = M - blockRow;
        if (zs < 128) {
            for (int z = t; z < (128 - zs) * 64; z += 256)
                As[zs + (z >> 6)][z & 63] = (_Float16)0.0f;
        }
    }

    v8f acc[4] = {};

    for (int k0 = 0; k0 < K; k0 += 64) {
        if (SRC16) {
            // ---- async f16 rows -> LDS (1024 chunks of 16B; 4 per thread) --
            #pragma unroll
            for (int i = 0; i < 4; ++i) {
                int idx  = i * 256 + t;
                int r    = idx >> 3, seg = idx & 7;   // 8 x 16B per 64-half row
                int grow = blockRow + r;
                if (grow < M) {
                    const _Float16* gp;
                    if (AMODE == A_DIRECT) {
                        gp = Xh + (long)grow * ldx + k0 + seg * 8;
                    } else if (AMODE == A_GATHER) {
                        gp = Xh + (long)gidx[grow] * ldx + k0 + seg * 8;
                    } else {  // A_FEAT: [e | e[rev] | h[src] | h[dst]]
                        int kc = k0 >> 6;
                        const _Float16* P; int rr;
                        if (kc == 0)      { P = fe16; rr = grow; }
                        else if (kc == 1) { P = fe16; rr = revmap[grow]; }
                        else if (kc == 2) { P = fh16; rr = src[grow]; }
                        else              { P = fh16; rr = dst[grow]; }
                        gp = P + (long)rr * DD + seg * 8;
                    }
                    async_ld16(as_base + r * 144 + seg * 16, gp);
                }
            }
        } else {
            // ---- vectorized f32 loads (float4) -> f16 LDS ----
            #pragma unroll
            for (int i = 0; i < 8; ++i) {
                int idx  = i * 256 + t;              // 2048 groups of 4 floats
                int r    = idx >> 4, c4 = (idx & 15) * 4;
                int grow = blockRow + r;
                int k    = k0 + c4;
                float4 v = make_float4(0.f, 0.f, 0.f, 0.f);
                if (grow < M && k < K)
                    v = *(const float4*)&X[(long)grow * ldx + k];
                v4h hv = { (_Float16)v.x, (_Float16)v.y,
                           (_Float16)v.z, (_Float16)v.w };
                *(v4h*)&As[r][c4] = hv;
            }
        }

        // ---- W tile: float4 global reads, transposed f16 stores ----
        #pragma unroll
        for (int i = 0; i < 4; ++i) {
            int idx = i * 256 + t;                   // 1024 groups of 4
            int k   = idx >> 4, n4 = (idx & 15) * 4;
            float4 wv = make_float4(0.f, 0.f, 0.f, 0.f);
            if (k0 + k < K)
                wv = *(const float4*)&W[(long)(k0 + k) * ldw + n4];
            Wt[n4 + 0][k] = (_Float16)wv.x;
            Wt[n4 + 1][k] = (_Float16)wv.y;
            Wt[n4 + 2][k] = (_Float16)wv.z;
            Wt[n4 + 3][k] = (_Float16)wv.w;
        }

        if (SRC16) async_wait0();
        __syncthreads();

        // ---- 2 K-steps of 32 per 64-wide chunk ----
        #pragma unroll
        for (int ks = 0; ks < 2; ++ks) {
            // A fragment 16x32: lane L<16 -> row L, K {0..7,16..23};
            // lane L>=16 -> row L-16, K {8..15,24..31}
            int ml   = lane & 15;
            int koff = (lane < 16) ? 0 : 8;
            v8h a0 = *(const v8h*)&As[m0 + ml][ks * 32 + koff];
            v8h a1 = *(const v8h*)&As[m0 + ml][ks * 32 + koff + 16];
            v16h a;
            #pragma unroll
            for (int i = 0; i < 8; ++i) { a[i] = a0[i]; a[i + 8] = a1[i]; }

            // B fragment 32x16: lane = N (mod 16); lanes<16 carry K {0..7,16..23},
            // lanes>=16 carry K {8..15,24..31} (same order as A)
            int kb = (lane < 16) ? 0 : 8;
            #pragma unroll
            for (int tt = 0; tt < 4; ++tt) {
                v8h b0 = *(const v8h*)&Wt[tt * 16 + (lane & 15)][ks * 32 + kb];
                v8h b1 = *(const v8h*)&Wt[tt * 16 + (lane & 15)][ks * 32 + kb + 16];
                v16h b;
                #pragma unroll
                for (int i = 0; i < 8; ++i) { b[i] = b0[i]; b[i + 8] = b1[i]; }
                acc[tt] = __builtin_amdgcn_wmma_f32_16x16x32_f16(
                    false, a, false, b, (short)0, acc[tt], false, false);
            }
        }
        __syncthreads();
    }

    // ---- fused BN stats scratch: reuse the A tile as float scratch ----
    float* sred = (float*)&As[0][0];
    if (STATS) {
        if (t < 128) sred[t] = 0.0f;
        __syncthreads();
    }

    // ---- epilogue: C/D layout -> lane (0..15)=N, VGPR r = M (lanes>=16: M+8)
    const int nl = lane & 15;
    const int mh = (lane >> 4) * 8;
    #pragma unroll
    for (int tt = 0; tt < 4; ++tt) {
        int   c  = tt * 16 + nl;
        float bv = bias[c];
        float psum = 0.0f, psq = 0.0f;
        #pragma unroll
        for (int r = 0; r < 8; ++r) {
            int grow = blockRow + m0 + mh + r;
            if (grow < M) {
                float v = acc[tt][r] + bv;
                if (GATHER2) v += G0[(long)src[grow] * DD + c]
                               +  G1[(long)dst[grow] * DD + c];
                if (ACCUM)   v += Y[(long)grow * ldy + c];
                if (RELU)    v = fmaxf(v, 0.0f);
                Y[(long)grow * ldy + c] = v;
                if (WRITE16) Y16[(long)grow * ldy + c] = (_Float16)v;
                if (STATS)   { psum += v; psq += v * v; }
            }
        }
        if (STATS) {
            atomicAdd(&sred[c], psum);
            atomicAdd(&sred[64 + c], psq);
        }
    }
    if (STATS) {
        __syncthreads();
        if (t < 128) atomicAdd(&statsOut[t], sred[t]);
    }
}

// ---------------------------------------------------------------------------
// Support kernels
// ---------------------------------------------------------------------------
__global__ void zero_kernel(float* __restrict__ p, long n) {
    long i = (long)blockIdx.x * blockDim.x + threadIdx.x;
    long stride = (long)gridDim.x * blockDim.x;
    for (; i < n; i += stride) p[i] = 0.0f;
}

// sig = sigmoid(e_hat); num[dst] += sig * Bh[src]; den[dst] += sig
// Reads the f16 shadow of e_hat; 2 columns per thread from one 4B load.
__global__ void aggregate_kernel(const _Float16* __restrict__ ehat16,
                                 const float* __restrict__ Bh,
                                 const int* __restrict__ src,
                                 const int* __restrict__ dst,
                                 float* __restrict__ num,
                                 float* __restrict__ den) {
    long i = (long)blockIdx.x * 256 + threadIdx.x;   // over EE * 32
    if (i >= (long)EE * 32) return;
    int k  = (int)(i >> 5);
    int c2 = (int)(i & 31) * 2;
    v2h hv = *(const v2h*)&ehat16[(long)k * DD + c2];
    float s0 = 1.0f / (1.0f + __expf(-(float)hv[0]));
    float s1 = 1.0f / (1.0f + __expf(-(float)hv[1]));
    long nb = (long)dst[k] * DD + c2;
    long sb = (long)src[k] * DD + c2;
    atomicAdd(&num[nb + 0], s0 * Bh[sb + 0]);
    atomicAdd(&num[nb + 1], s1 * Bh[sb + 1]);
    atomicAdd(&den[nb + 0], s0);
    atomicAdd(&den[nb + 1], s1);
}

// h_hat = Ah + num/(den+eps), and accumulate per-column sum/sumsq into stats
__global__ void hhat_stats_kernel(const float* __restrict__ Ah,
                                  const float* __restrict__ num,
                                  const float* __restrict__ den,
                                  float* __restrict__ h_hat,
                                  float* __restrict__ stats, long total) {
    __shared__ float ssum[64], ssq[64];
    int t = threadIdx.x;
    if (t < 64) { ssum[t] = 0.0f; ssq[t] = 0.0f; }
    __syncthreads();
    long i = (long)blockIdx.x * 256 + t;
    if (i < total) {
        int c = (int)(i & 63);
        float v = Ah[i] + num[i] / (den[i] + 1e-6f);
        h_hat[i] = v;
        atomicAdd(&ssum[c], v);
        atomicAdd(&ssq[c], v * v);
    }
    __syncthreads();
    if (t < 64) {
        atomicAdd(&stats[t],      ssum[t]);
        atomicAdd(&stats[64 + t], ssq[t]);
    }
}

// base += relu(bn(X)); also refresh the f16 shadow of base
__global__ void bn_apply_kernel(const float* __restrict__ X,
                                const float* __restrict__ stats,
                                const float* __restrict__ g,
                                const float* __restrict__ b,
                                float* __restrict__ base,
                                _Float16* __restrict__ base16,
                                long total, float invM) {
    long i = (long)blockIdx.x * 256 + threadIdx.x;
    if (i >= total) return;
    int c = (int)(i & 63);
    float mu  = stats[c] * invM;
    float var = stats[64 + c] * invM - mu * mu;
    float v = (X[i] - mu) * rsqrtf(var + 1e-5f) * g[c] + b[c];
    float nb = base[i] + fmaxf(v, 0.0f);
    base[i] = nb;
    base16[i] = (_Float16)nb;
}

// out[E x 2] = x1[E x 64] @ w2[64 x 2] + b2
__global__ void out_kernel(const float* __restrict__ x1,
                           const float* __restrict__ w2,
                           const float* __restrict__ b2,
                           float* __restrict__ out) {
    int r = blockIdx.x * 256 + threadIdx.x;
    if (r >= EE) return;
    float a0 = b2[0], a1 = b2[1];
    const float* xr = x1 + (long)r * DD;
    #pragma unroll
    for (int k = 0; k < DD; ++k) {
        float x = xr[k];
        a0 += x * w2[k * 2 + 0];
        a1 += x * w2[k * 2 + 1];
    }
    out[(long)r * 2 + 0] = a0;
    out[(long)r * 2 + 1] = a1;
}

// ---------------------------------------------------------------------------
// Host-side orchestration
// ---------------------------------------------------------------------------
extern "C" void kernel_launch(void* const* d_in, const int* in_sizes, int n_in,
                              void* d_out, int out_size, void* d_ws, size_t ws_size,
                              hipStream_t stream) {
    (void)in_sizes; (void)n_in; (void)out_size; (void)ws_size;

    const float* h_in    = (const float*)d_in[0];
    const float* e_in    = (const float*)d_in[1];
    const int*   src     = (const int*)  d_in[2];
    const int*   dst     = (const int*)  d_in[3];
    const int*   revmap  = (const int*)  d_in[4];
    const float* emb_h_w = (const float*)d_in[5];
    const float* emb_h_b = (const float*)d_in[6];
    const float* emb_e_w = (const float*)d_in[7];
    const float* emb_e_b = (const float*)d_in[8];
    const float* Aw  = (const float*)d_in[9];   const float* Abp = (const float*)d_in[10];
    const float* Bw  = (const float*)d_in[11];  const float* Bbp = (const float*)d_in[12];
    const float* Cw  = (const float*)d_in[13];  const float* Cbp = (const float*)d_in[14];
    const float* Dw  = (const float*)d_in[15];  const float* Dbp = (const float*)d_in[16];
    const float* Ew  = (const float*)d_in[17];  const float* Ebp = (const float*)d_in[18];
    const float* Fw  = (const float*)d_in[19];  const float* Fbp = (const float*)d_in[20];
    const float* Gw  = (const float*)d_in[21];  const float* Gbp = (const float*)d_in[22];
    const float* bn_h_g = (const float*)d_in[23]; const float* bn_h_b = (const float*)d_in[24];
    const float* bn_e_g = (const float*)d_in[25]; const float* bn_e_b = (const float*)d_in[26];
    const float* bn_u_g = (const float*)d_in[27]; const float* bn_u_b = (const float*)d_in[28];
    const float* mlp_w0 = (const float*)d_in[29]; const float* mlp_b0 = (const float*)d_in[30];
    const float* mlp_w1 = (const float*)d_in[31]; const float* mlp_b1 = (const float*)d_in[32];
    const float* mlp_w2 = (const float*)d_in[33]; const float* mlp_b2 = (const float*)d_in[34];

    const long NH = (long)NN * DD;
    const long NE = (long)EE * DD;

    // f32 workspace
    float* p     = (float*)d_ws;
    float* h     = p; p += NH;
    float* e     = p; p += NE;
    float* u     = p; p += NE;
    float* u_hat = p; p += NE;
    float* e_hat = p; p += NE;
    float* Ah    = p; p += NH;
    float* Bh    = p; p += NH;
    float* Dh    = p; p += NH;
    float* Eh    = p; p += NH;
    float* num   = p; p += NH;
    float* den   = p; p += NH;
    float* h_hat = p; p += NH;
    float* stats_h = p; p += 256;     // stats_h/e/u contiguous (one zero pass)
    float* stats_e = p; p += 256;
    float* stats_u = p; p += 256;
    // f16 shadows (16B-aligned: everything above is multiples of 64 floats)
    _Float16* q      = (_Float16*)p;
    _Float16* h16    = q; q += NH;
    _Float16* e16    = q; q += NE;
    _Float16* u16    = q; q += NE;     // + ehat16 = contiguous 2*NE region
    _Float16* ehat16 = q; q += NE;
    // readout overlays (layer buffers dead by then)
    float*    x0    = u;               // 400000 x 128 f32, overlays [u, u_hat]
    _Float16* x016  = u16;             // 400000 x 128 f16, overlays [u16, ehat16]
    float*    x1    = e_hat;           // 400000 x 64 f32

    const dim3 blk(256);
    const int gridNodes = (NN + 127) / 128;
    const int gridEdges = (EE + 127) / 128;
    const int gridNHel  = (int)((NH + 255) / 256);
    const int gridNEel  = (int)((NE + 255) / 256);
    const int gridAgg   = (int)(((long)EE * 32 + 255) / 256);

    // ---- embeddings (f32 inputs): also produce f16 shadows; u = 0 ----
    gemm64_kernel<A_DIRECT, false, false, false, false, true, false>
        <<<gridNodes, blk, 0, stream>>>(
        h_in, 64, nullptr, nullptr, nullptr, nullptr, nullptr, nullptr, nullptr,
        emb_h_w, DD, emb_h_b, nullptr, nullptr, h, DD, h16, nullptr, NN, 64);
    gemm64_kernel<A_DIRECT, false, false, false, false, true, false>
        <<<gridEdges, blk, 0, stream>>>(
        e_in, 16, nullptr, nullptr, nullptr, nullptr, nullptr, nullptr, nullptr,
        emb_e_w, DD, emb_e_b, nullptr, nullptr, e, DD, e16, nullptr, EE, 16);
    zero_kernel<<<2048, blk, 0, stream>>>(u, NE);
    zero_kernel<<<1024, blk, 0, stream>>>((float*)u16, NE / 2);

    // ---- gated-GCN layers ----
    for (int l = 0; l < NLAYERS; ++l) {
        const long wo = (long)l * DD * DD;
        const long bo = (long)l * DD;

        zero_kernel<<<1024, blk, 0, stream>>>(num, NH);
        zero_kernel<<<1024, blk, 0, stream>>>(den, NH);
        zero_kernel<<<1, blk, 0, stream>>>(stats_h, 768);   // h/e/u stats

        // node projections Ah, Bh, Dh, Eh  (read h16, async path)
        gemm64_kernel<A_DIRECT, true, false, false, false, false, false>
            <<<gridNodes, blk, 0, stream>>>(
            nullptr, DD, h16, nullptr, nullptr, nullptr, nullptr, nullptr, nullptr,
            Aw + wo, DD, Abp + bo, nullptr, nullptr, Ah, DD, nullptr, nullptr, NN, DD);
        gemm64_kernel<A_DIRECT, true, false, false, false, false, false>
            <<<gridNodes, blk, 0, stream>>>(
            nullptr, DD, h16, nullptr, nullptr, nullptr, nullptr, nullptr, nullptr,
            Bw + wo, DD, Bbp + bo, nullptr, nullptr, Bh, DD, nullptr, nullptr, NN, DD);
        gemm64_kernel<A_DIRECT, true, false, false, false, false, false>
            <<<gridNodes, blk, 0, stream>>>(
            nullptr, DD, h16, nullptr, nullptr, nullptr, nullptr, nullptr, nullptr,
            Dw + wo, DD, Dbp + bo, nullptr, nullptr, Dh, DD, nullptr, nullptr, NN, DD);
        gemm64_kernel<A_DIRECT, true, false, false, false, false, false>
            <<<gridNodes, blk, 0, stream>>>(
            nullptr, DD, h16, nullptr, nullptr, nullptr, nullptr, nullptr, nullptr,
            Ew + wo, DD, Ebp + bo, nullptr, nullptr, Eh, DD, nullptr, nullptr, NN, DD);

        // e_hat = e @ C + Cb + Dh[src] + Eh[dst]; f16 shadow + fused BN stats
        gemm64_kernel<A_DIRECT, true, false, false, true, true, true>
            <<<gridEdges, blk, 0, stream>>>(
            nullptr, DD, e16, nullptr, nullptr, nullptr, src, dst, nullptr,
            Cw + wo, DD, Cbp + bo, Dh, Eh, e_hat, DD, ehat16, stats_e, EE, DD);

        // u_hat = u @ F + Fb;  u_hat += e_hat[revmap] @ G + Gb  (+ fused stats)
        gemm64_kernel<A_DIRECT, true, false, false, false, false, false>
            <<<gridEdges, blk, 0, stream>>>(
            nullptr, DD, u16, nullptr, nullptr, nullptr, nullptr, nullptr, nullptr,
            Fw + wo, DD, Fbp + bo, nullptr, nullptr, u_hat, DD, nullptr, nullptr, EE, DD);
        gemm64_kernel<A_GATHER, true, false, true, false, false, true>
            <<<gridEdges, blk, 0, stream>>>(
            nullptr, DD, ehat16, revmap, nullptr, nullptr, nullptr, nullptr, nullptr,
            Gw + wo, DD, Gbp + bo, nullptr, nullptr, u_hat, DD, nullptr, stats_u, EE, DD);

        // gated aggregation into num/den (reads ehat16)
        aggregate_kernel<<<gridAgg, blk, 0, stream>>>(ehat16, Bh, src, dst, num, den);

        // h update: h += relu(bn(Ah + num/(den+eps)))
        hhat_stats_kernel<<<gridNHel, blk, 0, stream>>>(Ah, num, den, h_hat, stats_h, NH);
        bn_apply_kernel<<<gridNHel, blk, 0, stream>>>(
            h_hat, stats_h, bn_h_g + bo, bn_h_b + bo, h, h16, NH, 1.0f / (float)NN);

        // e update
        bn_apply_kernel<<<gridNEel, blk, 0, stream>>>(
            e_hat, stats_e, bn_e_g + bo, bn_e_b + bo, e, e16, NE, 1.0f / (float)EE);

        // u update
        bn_apply_kernel<<<gridNEel, blk, 0, stream>>>(
            u_hat, stats_u, bn_u_g + bo, bn_u_b + bo, u, u16, NE, 1.0f / (float)EE);
    }

    // ---- readout: feat = [e | e[rev] | h[src] | h[dst]] (never materialized)
    gemm64_kernel<A_FEAT, true, true, false, false, true, false>
        <<<gridEdges, blk, 0, stream>>>(
        nullptr, 0, nullptr, nullptr, e16, h16, src, dst, revmap,
        mlp_w0, 128, mlp_b0, nullptr, nullptr, x0, 128, x016, nullptr, EE, 256);
    gemm64_kernel<A_FEAT, true, true, false, false, true, false>
        <<<gridEdges, blk, 0, stream>>>(
        nullptr, 0, nullptr, nullptr, e16, h16, src, dst, revmap,
        mlp_w0 + 64, 128, mlp_b0 + 64, nullptr, nullptr, x0 + 64, 128, x016 + 64,
        nullptr, EE, 256);
    // x1 = relu(x0 @ w1 + b1)   (reads x0's f16 shadow)
    gemm64_kernel<A_DIRECT, true, true, false, false, false, false>
        <<<gridEdges, blk, 0, stream>>>(
        nullptr, 128, x016, nullptr, nullptr, nullptr, nullptr, nullptr, nullptr,
        mlp_w1, DD, mlp_b1, nullptr, nullptr, x1, DD, nullptr, nullptr, EE, 128);
    // out = x1 @ w2 + b2
    out_kernel<<<(EE + 255) / 256, blk, 0, stream>>>(x1, mlp_w2, mlp_b2, (float*)d_out);
}
